// VanillaMPN2_12214886990604
// MI455X (gfx1250) — compile-verified, matrix operand order checked by reading, not run on hardware
//
#include <hip/hip_runtime.h>

// ---------------------------------------------------------------------------
// CDNA5 (gfx1250) implementation of VanillaMPN (4-step message-passing GNN).
// wave32 / WMMA f16->f32. Activations stored f16, BN stats fused into GEMM
// epilogues, scatter-add via global f32 atomics. Weights pre-transposed and
// converted to f16 once per launch so per-block B staging is a b128 stream.
// ---------------------------------------------------------------------------

typedef __attribute__((ext_vector_type(16))) _Float16 v16h;
typedef __attribute__((ext_vector_type(8)))  float    v8f;

#define TILE_M 128
#define TB 256

// Load a 16x32 f16 fragment (A-layout per CDNA5 ISA 7.12.2) from an LDS tile
// stored row-major [rows][stride]. For B we load W^T with the same layout
// (lane = output column, K runs along the register pattern).
__device__ __forceinline__ v16h load_frag16x32(const _Float16* __restrict__ tile,
                                               int stride, int row0, int k0) {
    int lane = threadIdx.x & 31;
    const _Float16* p = tile + (row0 + (lane & 15)) * stride + k0 + ((lane >> 4) << 3);
    v16h f;
#pragma unroll
    for (int d = 0; d < 4; ++d) {
        f[2 * d]         = p[2 * d];
        f[2 * d + 1]     = p[2 * d + 1];
        f[8 + 2 * d]     = p[16 + 2 * d];
        f[8 + 2 * d + 1] = p[16 + 2 * d + 1];
    }
    return f;
}

// Pre-transpose one weight matrix: WT[n*Kpad + k] = (f16)W[k*128 + n], zero pad.
__global__ void transpose_w(const float* __restrict__ W, _Float16* __restrict__ WT,
                            int Kreal, int Kpad) {
    int idx = blockIdx.x * blockDim.x + threadIdx.x;
    if (idx >= 128 * Kpad) return;
    int n = idx / Kpad, k = idx - n * Kpad;
    WT[idx] = (_Float16)((k < Kreal) ? W[(long)k * 128 + n] : 0.f);
}

// AMODE: 0 = plain f32 source (rows x Kreal, zero-padded to Kpad)
//        1 = f16 source with fused BN affine (K = 128)
//        2 = gather-concat [node[i] | node[j] | edge]   (K = 384)
//        3 = gather-concat [node[i] | edge]             (K = 256)
template <int AMODE, bool RELU, bool STATS>
__global__ void __launch_bounds__(TB)
gemm_kernel(const float* __restrict__ A32, const _Float16* __restrict__ A16,
            const _Float16* __restrict__ nodeF, const _Float16* __restrict__ edgeF,
            const int* __restrict__ idxI, const int* __restrict__ idxJ,
            const float* __restrict__ scaleA, const float* __restrict__ shiftA,
            const _Float16* __restrict__ WT, const float* __restrict__ bias,
            _Float16* __restrict__ out, float* __restrict__ gsum, float* __restrict__ gsq,
            long Mrows, int Kreal, int Kpad)
{
    extern __shared__ char smemRaw[];
    _Float16* ldsA = (_Float16*)smemRaw;                 // [TILE_M][Kpad] f16
    _Float16* ldsB = ldsA + TILE_M * Kpad;               // [128][Kpad] f16 (W^T)
    float* ldsStats = (float*)(ldsB + 128 * Kpad);       // [256] f32

    const int tid = threadIdx.x;
    const long rowBase = (long)blockIdx.x * TILE_M;

    if (STATS && tid < 256) ldsStats[tid] = 0.f;

    // ---- stage A tile into LDS (f16, zero-padded) ----
    const int rowSegs = Kpad >> 3;  // 8-half (16B) segments per row
    for (int idx = tid; idx < TILE_M * rowSegs; idx += TB) {
        int r = idx / rowSegs;
        int k = (idx - r * rowSegs) << 3;
        long row  = rowBase + r;
        long crow = row < Mrows ? row : (Mrows - 1);
        if (AMODE == 0) {
            _Float16 tmp[8];
#pragma unroll
            for (int t = 0; t < 8; ++t) {
                int kk = k + t;
                tmp[t] = (_Float16)((kk < Kreal) ? A32[crow * (long)Kreal + kk] : 0.f);
            }
            *(int4*)&ldsA[r * Kpad + k] = *(int4*)tmp;
        } else if (AMODE == 1) {
            _Float16 tmp[8];
            const _Float16* src = A16 + crow * 128 + k;
#pragma unroll
            for (int t = 0; t < 8; ++t)
                tmp[t] = (_Float16)((float)src[t] * scaleA[k + t] + shiftA[k + t]);
            *(int4*)&ldsA[r * Kpad + k] = *(int4*)tmp;
        } else {
            const _Float16* src;
            if (AMODE == 2) {
                if (k < 128)      src = nodeF + (long)idxI[crow] * 128 + k;
                else if (k < 256) src = nodeF + (long)idxJ[crow] * 128 + (k - 128);
                else              src = edgeF + crow * 128 + (k - 256);
            } else {
                if (k < 128)      src = nodeF + (long)idxI[crow] * 128 + k;
                else              src = edgeF + crow * 128 + (k - 128);
            }
            *(int4*)&ldsA[r * Kpad + k] = *(const int4*)src;
        }
    }

    // ---- stage B = pre-transposed f16 W^T: straight b128 copy ----
    {
        const int bSegs = (128 * Kpad) >> 3;
        for (int idx = tid; idx < bSegs; idx += TB)
            *(int4*)&ldsB[idx << 3] = *(const int4*)&WT[idx << 3];
    }

    __syncthreads();

    // 8 waves arranged 4x2: each wave owns a 32x64 output tile
    // (2 A-frags + 4 B-frags per K-chunk -> 6 LDS frag loads per 8 WMMAs).
    const int wave = tid >> 5;
    const int lane = tid & 31;
    const int mb = (wave & 3) * 32;
    const int nb = (wave >> 2) * 64;

    v8f acc[2][4];
#pragma unroll
    for (int i = 0; i < 2; ++i)
#pragma unroll
        for (int j = 0; j < 4; ++j)
#pragma unroll
            for (int q = 0; q < 8; ++q) acc[i][j][q] = 0.f;

    for (int k0 = 0; k0 < Kpad; k0 += 32) {
        v16h a0 = load_frag16x32(ldsA, Kpad, mb, k0);
        v16h a1 = load_frag16x32(ldsA, Kpad, mb + 16, k0);
#pragma unroll
        for (int j = 0; j < 4; ++j) {
            v16h b = load_frag16x32(ldsB, Kpad, nb + j * 16, k0);
            acc[0][j] = __builtin_amdgcn_wmma_f32_16x16x32_f16(
                false, a0, false, b, (short)0, acc[0][j], false, false);
            acc[1][j] = __builtin_amdgcn_wmma_f32_16x16x32_f16(
                false, a1, false, b, (short)0, acc[1][j], false, false);
        }
    }

    // ---- epilogue: bias (+ReLU), store f16, fused BN batch statistics ----
    const int half = lane >> 4;
    const int ncol = lane & 15;
#pragma unroll
    for (int j = 0; j < 4; ++j) {
        int col = nb + j * 16 + ncol;
        float bv = bias[col];
        float s0 = 0.f, s1 = 0.f;
#pragma unroll
        for (int i = 0; i < 2; ++i) {
#pragma unroll
            for (int r = 0; r < 8; ++r) {
                long row = rowBase + mb + i * 16 + r + 8 * half;  // D-matrix layout
                float y = acc[i][j][r] + bv;
                if (RELU) y = fmaxf(y, 0.f);
                if (row < Mrows) {
                    out[row * 128 + col] = (_Float16)y;
                    if (STATS) { s0 += y; s1 += y * y; }
                }
            }
        }
        if (STATS) {
            atomicAdd(&ldsStats[col], s0);        // ds_add_f32
            atomicAdd(&ldsStats[128 + col], s1);
        }
    }
    if (STATS) {
        __syncthreads();
        if (tid < 128) {
            atomicAdd(&gsum[tid], ldsStats[tid]);       // global_atomic_add_f32
            atomicAdd(&gsq[tid],  ldsStats[128 + tid]);
        }
    }
}

// ---- small helper kernels -------------------------------------------------

__global__ void zero_f32(float* p, long n) {
    long i = (long)blockIdx.x * blockDim.x + threadIdx.x;
    if (i < n) p[i] = 0.f;
}

__global__ void zero_stats(float* gsum, float* gsq) {
    int i = threadIdx.x;
    gsum[i] = 0.f; gsq[i] = 0.f;
}

__global__ void bn_finalize(const float* __restrict__ gsum, const float* __restrict__ gsq,
                            const float* __restrict__ gamma, const float* __restrict__ beta,
                            float* __restrict__ scale, float* __restrict__ shift,
                            float invCount) {
    int c = threadIdx.x;  // 128
    float m = gsum[c] * invCount;
    float v = gsq[c] * invCount - m * m;
    float s = gamma[c] * rsqrtf(v + 1e-5f);
    scale[c] = s;
    shift[c] = beta[c] - m * s;
}

__global__ void apply_bn(const _Float16* __restrict__ h, const float* __restrict__ scale,
                         const float* __restrict__ shift, _Float16* __restrict__ out, long total) {
    long i = (long)blockIdx.x * blockDim.x + threadIdx.x;
    if (i >= total) return;
    int c = (int)(i & 127);
    out[i] = (_Float16)((float)h[i] * scale[c] + shift[c]);
}

__global__ void scatter_add(const _Float16* __restrict__ h, const float* __restrict__ scale,
                            const float* __restrict__ shift, const int* __restrict__ idxI,
                            float* __restrict__ acc, long E) {
    long i = (long)blockIdx.x * blockDim.x + threadIdx.x;  // E*32 threads, 4 cols each
    if (i >= E * 32) return;
    long e  = i >> 5;
    int  c0 = (int)(i & 31) * 4;
    long n  = idxI[e];
    const _Float16* src = h + e * 128 + c0;
    float* dst = acc + n * 128 + c0;
#pragma unroll
    for (int t = 0; t < 4; ++t)
        atomicAdd(&dst[t], (float)src[t] * scale[c0 + t] + shift[c0 + t]);
}

__global__ void node_commit(float* __restrict__ acc, _Float16* __restrict__ nodeF, long n) {
    long i = (long)blockIdx.x * blockDim.x + threadIdx.x;
    if (i < n) { nodeF[i] = (_Float16)acc[i]; acc[i] = 0.f; }
}

__global__ void head_kernel(const _Float16* __restrict__ edgeF, const float* __restrict__ Wc,
                            const float* __restrict__ bc, float* __restrict__ out, long E) {
    long e = (long)blockIdx.x * 8 + (threadIdx.x >> 5);  // one wave32 per edge
    int lane = threadIdx.x & 31;
    if (e >= E) return;
    const _Float16* row = edgeF + e * 128;
    float s = 0.f;
#pragma unroll
    for (int t = 0; t < 4; ++t) {
        int c = lane * 4 + t;
        s += (float)row[c] * Wc[c];
    }
#pragma unroll
    for (int off = 16; off; off >>= 1) s += __shfl_xor(s, off, 32);
    if (lane == 0) out[e] = s + bc[0];
}

// ---------------------------------------------------------------------------

extern "C" void kernel_launch(void* const* d_in, const int* in_sizes, int n_in,
                              void* d_out, int out_size, void* d_ws, size_t ws_size,
                              hipStream_t stream)
{
    const float* x     = (const float*)d_in[0];
    const float* eattr = (const float*)d_in[1];
    const int*   eidx  = (const int*)d_in[2];
    const float* nW1 = (const float*)d_in[3];  const float* nb1  = (const float*)d_in[4];
    const float* ng1 = (const float*)d_in[5];  const float* nbe1 = (const float*)d_in[6];
    const float* nW2 = (const float*)d_in[7];  const float* nb2  = (const float*)d_in[8];
    const float* eW1 = (const float*)d_in[9];  const float* eb1  = (const float*)d_in[10];
    const float* eg1 = (const float*)d_in[11]; const float* ebe1 = (const float*)d_in[12];
    const float* eW2 = (const float*)d_in[13]; const float* eb2  = (const float*)d_in[14];
    const float* We  = (const float*)d_in[15]; const float* be   = (const float*)d_in[16];
    const float* ge  = (const float*)d_in[17]; const float* bee  = (const float*)d_in[18];
    const float* Wn  = (const float*)d_in[19]; const float* bnn  = (const float*)d_in[20];
    const float* gn  = (const float*)d_in[21]; const float* ben  = (const float*)d_in[22];
    const float* Wc  = (const float*)d_in[23]; const float* bc   = (const float*)d_in[24];

    const long N = in_sizes[0] / 128;   // x: [N,128]
    const long E = in_sizes[1] / 291;   // edge_attr: [E,291]
    const int* idxJ = eidx;             // edge_index[0] = source j
    const int* idxI = eidx + E;         // edge_index[1] = target i

    // workspace layout
    char* ws = (char*)d_ws;
    auto align256 = [](size_t v) { return (v + 255) & ~(size_t)255; };
    size_t off = 0;
    _Float16* nodeF = (_Float16*)(ws + off); off = align256(off + (size_t)N * 128 * 2);
    _Float16* edgeF = (_Float16*)(ws + off); off = align256(off + (size_t)E * 128 * 2);
    _Float16* h16   = (_Float16*)(ws + off); off = align256(off + (size_t)E * 128 * 2);
    float* nodeAcc  = (float*)(ws + off);    off = align256(off + (size_t)N * 128 * 4);
    float* gsum     = (float*)(ws + off);    off += 128 * 4;
    float* gsq      = (float*)(ws + off);    off += 128 * 4;
    float* bnScale  = (float*)(ws + off);    off += 128 * 4;
    float* bnShift  = (float*)(ws + off);    off = align256(off + 128 * 4);
    // pre-transposed f16 weights
    _Float16* wtN1 = (_Float16*)(ws + off); off = align256(off + (size_t)128 * 128 * 2);
    _Float16* wtN2 = (_Float16*)(ws + off); off = align256(off + (size_t)128 * 128 * 2);
    _Float16* wtE1 = (_Float16*)(ws + off); off = align256(off + (size_t)128 * 320 * 2);
    _Float16* wtE2 = (_Float16*)(ws + off); off = align256(off + (size_t)128 * 128 * 2);
    _Float16* wtMPe[4];
    for (int s = 0; s < 4; ++s) { wtMPe[s] = (_Float16*)(ws + off); off = align256(off + (size_t)128 * 384 * 2); }
    _Float16* wtMPn[4];
    for (int s = 0; s < 4; ++s) { wtMPn[s] = (_Float16*)(ws + off); off = align256(off + (size_t)128 * 256 * 2); }

    const int nBlkN = (int)((N + TILE_M - 1) / TILE_M);
    const int nBlkE = (int)((E + TILE_M - 1) / TILE_M);
    auto ldsBytes = [](int Kpad) { return (size_t)512 * Kpad + 1024; };
    auto twBlocks = [](int Kpad) { return (128 * Kpad + 255) / 256; };

    // ---- one-time per launch: pre-transpose + f16-convert all weights ----
    transpose_w<<<twBlocks(128), 256, 0, stream>>>(nW1, wtN1, 128, 128);
    transpose_w<<<twBlocks(128), 256, 0, stream>>>(nW2, wtN2, 128, 128);
    transpose_w<<<twBlocks(320), 256, 0, stream>>>(eW1, wtE1, 291, 320);
    transpose_w<<<twBlocks(128), 256, 0, stream>>>(eW2, wtE2, 128, 128);
    for (int s = 0; s < 4; ++s) {
        transpose_w<<<twBlocks(384), 256, 0, stream>>>(We + (size_t)s * 384 * 128, wtMPe[s], 384, 384);
        transpose_w<<<twBlocks(256), 256, 0, stream>>>(Wn + (size_t)s * 256 * 128, wtMPn[s], 256, 256);
    }

    // zero node accumulator (deterministic each call)
    {
        long total = N * 128;
        zero_f32<<<(int)((total + 255) / 256), 256, 0, stream>>>(nodeAcc, total);
    }

    // ---- node embedding MLP: BN(relu(x@W1+b1)) @ W2 + b2 ----
    zero_stats<<<1, 128, 0, stream>>>(gsum, gsq);
    gemm_kernel<0, true, true><<<nBlkN, TB, ldsBytes(128), stream>>>(
        x, nullptr, nullptr, nullptr, nullptr, nullptr, nullptr, nullptr,
        wtN1, nb1, h16, gsum, gsq, N, 128, 128);
    bn_finalize<<<1, 128, 0, stream>>>(gsum, gsq, ng1, nbe1, bnScale, bnShift, 1.f / (float)N);
    gemm_kernel<1, false, false><<<nBlkN, TB, ldsBytes(128), stream>>>(
        nullptr, h16, nullptr, nullptr, nullptr, nullptr, bnScale, bnShift,
        wtN2, nb2, nodeF, nullptr, nullptr, N, 128, 128);

    // ---- edge embedding MLP (K = 291, padded to 320) ----
    zero_stats<<<1, 128, 0, stream>>>(gsum, gsq);
    gemm_kernel<0, true, true><<<nBlkE, TB, ldsBytes(320), stream>>>(
        eattr, nullptr, nullptr, nullptr, nullptr, nullptr, nullptr, nullptr,
        wtE1, eb1, h16, gsum, gsq, E, 291, 320);
    bn_finalize<<<1, 128, 0, stream>>>(gsum, gsq, eg1, ebe1, bnScale, bnShift, 1.f / (float)E);
    gemm_kernel<1, false, false><<<nBlkE, TB, ldsBytes(128), stream>>>(
        nullptr, h16, nullptr, nullptr, nullptr, nullptr, bnScale, bnShift,
        wtE2, eb2, edgeF, nullptr, nullptr, E, 128, 128);

    // ---- 4 message-passing steps ----
    for (int s = 0; s < 4; ++s) {
        // edge MLP: BN(relu(cat(x_i, x_j, e) @ We + be)) -> edgeF
        zero_stats<<<1, 128, 0, stream>>>(gsum, gsq);
        gemm_kernel<2, true, true><<<nBlkE, TB, ldsBytes(384), stream>>>(
            nullptr, nullptr, nodeF, edgeF, idxI, idxJ, nullptr, nullptr,
            wtMPe[s], be + s * 128, h16, gsum, gsq, E, 384, 384);
        bn_finalize<<<1, 128, 0, stream>>>(gsum, gsq, ge + s * 128, bee + s * 128,
                                           bnScale, bnShift, 1.f / (float)E);
        {
            long total = E * 128;
            apply_bn<<<(int)((total + 255) / 256), 256, 0, stream>>>(h16, bnScale, bnShift,
                                                                     edgeF, total);
        }
        // message MLP: BN(relu(cat(x_i, edgeF) @ Wn + bn)) then scatter-add to i
        zero_stats<<<1, 128, 0, stream>>>(gsum, gsq);
        gemm_kernel<3, true, true><<<nBlkE, TB, ldsBytes(256), stream>>>(
            nullptr, nullptr, nodeF, edgeF, idxI, nullptr, nullptr, nullptr,
            wtMPn[s], bnn + s * 128, h16, gsum, gsq, E, 256, 256);
        bn_finalize<<<1, 128, 0, stream>>>(gsum, gsq, gn + s * 128, ben + s * 128,
                                           bnScale, bnShift, 1.f / (float)E);
        {
            long total = E * 32;
            scatter_add<<<(int)((total + 255) / 256), 256, 0, stream>>>(h16, bnScale, bnShift,
                                                                        idxI, nodeAcc, E);
        }
        {
            long total = N * 128;
            node_commit<<<(int)((total + 255) / 256), 256, 0, stream>>>(nodeAcc, nodeF, total);
        }
    }

    // ---- classification head: edgeF @ Wc + bc -> [E,1] f32 ----
    head_kernel<<<(int)((E + 7) / 8), TB, 0, stream>>>(edgeF, Wc, bc, (float*)d_out, E);
}
